// LoRALinearPerSubject_89489938579617
// MI455X (gfx1250) — compile-verified, matrix operand order checked by reading, not run on hardware
//
#include <hip/hip_runtime.h>

#define IN_F   1024
#define OUT_F  1024
#define RANKD  8
#define NADAPT 8
#define BATCH  32
#define SEQ    1024
#define MTOT   (BATCH * SEQ)
#define SCALE_F 0.125f

// GEMM tiling
#define BM 128
#define BN 256
#define BK 32
#define LDK 40                 // bf16 K-stride: 64B row + 16B pad (= TDM pad_interval 16DW / pad_amount 4DW)
#define NK (IN_F / BK)         // 32 k-iterations

typedef __attribute__((ext_vector_type(16))) __bf16 v16bf;
typedef __attribute__((ext_vector_type(8)))  float  v8f;
typedef __attribute__((ext_vector_type(4)))  unsigned int u32x4;
typedef __attribute__((ext_vector_type(8)))  int i32x8;
typedef __attribute__((ext_vector_type(4)))  int i32x4;

#if __has_builtin(__builtin_amdgcn_tensor_load_to_lds)
#define USE_TDM 1
#else
#define USE_TDM 0
#endif

__device__ __forceinline__ unsigned f2bf(float f) {
  union { float f; unsigned u; } v; v.f = f;
  unsigned r = v.u + 0x7FFFu + ((v.u >> 16) & 1u);   // round-to-nearest-even
  return r >> 16;
}

// ---------------------------------------------------------------------------
// weff[e][o][d] = bf16( W[o][d] + SCALE * sum_r Bw[e][o][r] * A[e][r][d] )
// ---------------------------------------------------------------------------
__global__ __launch_bounds__(256) void prep_weff(
    const float* __restrict__ W, const float* __restrict__ A,
    const float* __restrict__ Bw, unsigned short* __restrict__ weff)
{
  int t  = blockIdx.x * 256 + threadIdx.x;   // 8*1024*128 threads
  int d0 = (t & 127) * 8;
  int o  = (t >> 7) & 1023;
  int e  = t >> 17;

  const float* wrow = W + (size_t)o * IN_F + d0;
  float acc[8];
#pragma unroll
  for (int j = 0; j < 8; ++j) acc[j] = wrow[j];

  const float* brow = Bw + ((size_t)e * OUT_F + o) * RANKD;
  const float* arow = A + (size_t)e * RANKD * IN_F + d0;
#pragma unroll
  for (int r = 0; r < RANKD; ++r) {
    float s = brow[r] * SCALE_F;
#pragma unroll
    for (int j = 0; j < 8; ++j) acc[j] = fmaf(s, arow[(size_t)r * IN_F + j], acc[j]);
  }

  uint4 pkt;
  pkt.x = f2bf(acc[0]) | (f2bf(acc[1]) << 16);
  pkt.y = f2bf(acc[2]) | (f2bf(acc[3]) << 16);
  pkt.z = f2bf(acc[4]) | (f2bf(acc[5]) << 16);
  pkt.w = f2bf(acc[6]) | (f2bf(acc[7]) << 16);
  *(uint4*)(weff + ((size_t)e * OUT_F + o) * IN_F + d0) = pkt;
}

// ---------------------------------------------------------------------------
// xb = bf16(x), one pass (makes x TDM-eligible: TDM moves raw bytes only)
// ---------------------------------------------------------------------------
__global__ __launch_bounds__(256) void cvt_x(const float* __restrict__ x,
                                             unsigned short* __restrict__ xb)
{
  size_t t = (size_t)blockIdx.x * 256 + threadIdx.x;   // MTOT*IN_F/8 threads
  const float4* p = (const float4*)x + t * 2;
  float4 a = p[0], b = p[1];
  uint4 o;
  o.x = f2bf(a.x) | (f2bf(a.y) << 16);
  o.y = f2bf(a.z) | (f2bf(a.w) << 16);
  o.z = f2bf(b.x) | (f2bf(b.y) << 16);
  o.w = f2bf(b.z) | (f2bf(b.w) << 16);
  ((uint4*)xb)[t] = o;
}

#if USE_TDM
// Issue one TDM 2D tile load: rows x BK bf16 elements, row stride IN_F elems,
// into LDS with 16B pad after each 64B row (LDS row stride = LDK elements).
__device__ __forceinline__ void tdm_load_tile(unsigned ldsaddr,
                                              const unsigned short* gsrc,
                                              int rows)
{
  unsigned long long ga = (unsigned long long)(size_t)gsrc;
  u32x4 g0;
  g0[0] = 1u;                                   // count=1 (valid user descriptor)
  g0[1] = ldsaddr;                              // LDS byte address
  g0[2] = (unsigned)ga;                         // global_addr[31:0]
  g0[3] = (unsigned)(ga >> 32) | (2u << 30);    // global_addr[56:32] | type=2 (image)

  i32x8 g1;
  g1[0] = (int)((1u << 16)        // data_size = 2 bytes
              | (1u << 20)        // pad_enable
              | (3u << 22)        // pad_interval: 16 DWORDs (one 64B tile row)
              | (3u << 25));      // pad_amount: 4 DWORDs (16B)
  g1[1] = (int)(((unsigned)IN_F & 0xFFFFu) << 16);  // tensor_dim0[15:0]
  g1[2] = (int)(0u | (0u << 16));                   // tensor_dim0[31:16]=0, tensor_dim1[15:0]=0
  g1[3] = (int)(0x10u | ((unsigned)BK << 16));      // tensor_dim1[31:16]=0x10 (1M rows), tile_dim0=BK
  g1[4] = (int)((unsigned)rows);                    // tile_dim1=rows, tile_dim2=0
  g1[5] = (int)(unsigned)IN_F;                      // tensor_dim0_stride[31:0]
  g1[6] = 0;                                        // stride[47:32]=0, dim1_stride lo=0
  g1[7] = 0;                                        // dim1_stride hi=0

  i32x4 z4; z4[0] = 0; z4[1] = 0; z4[2] = 0; z4[3] = 0;
  i32x8 z8;
#pragma unroll
  for (int i = 0; i < 8; ++i) z8[i] = 0;
  // 6-arg form (clang-23 / therock-10.0 headers): (g0, g1, g2, g3, g4, cpol)
  __builtin_amdgcn_tensor_load_to_lds(g0, g1, z4, z4, z8, 0);
}

__device__ __forceinline__ unsigned lds_off(const void* p) {
  return (unsigned)(size_t)p;   // flat LDS address: low 32 bits == LDS byte offset
}
#endif

// ---------------------------------------------------------------------------
// out[m][n] = sum_k xb[m][k] * weff[sid][n][k] + bias[n]
// 256 threads (8 waves), block tile 128x256, wave tile 64x64 (4x4 WMMA bf16).
// A/B tiles staged global->LDS by the Tensor Data Mover (wave 0), double buffered.
// ---------------------------------------------------------------------------
__global__ __launch_bounds__(256) void lora_gemm(
    const unsigned short* __restrict__ xb, const unsigned short* __restrict__ weff,
    const float* __restrict__ bias, const int* __restrict__ sid,
    float* __restrict__ out)
{
  __shared__ __align__(16) unsigned short sA[2][BM * LDK];
  __shared__ __align__(16) unsigned short sB[2][BN * LDK];

  const int tid = threadIdx.x;
  const int m0  = blockIdx.x * BM;          // 128 rows, always within one batch (128 | 1024)
  const int n0  = blockIdx.y * BN;
  const int subj = sid[m0 >> 10];
  const unsigned short* wbase = weff + (size_t)subj * OUT_F * IN_F;

  const int lane = tid & 31;
  const int wid  = tid >> 5;
  const int wm   = wid >> 2;                // 0..1
  const int wn   = wid & 3;                 // 0..3
  const int lr   = lane & 15;               // row within 16x16 fragment
  const int lkb  = (lane >> 4) * 8;         // K sub-chunk: 0 or 8 elements

  v8f acc[4][4];
  {
    v8f zero;
#pragma unroll
    for (int k = 0; k < 8; ++k) zero[k] = 0.0f;
#pragma unroll
    for (int i = 0; i < 4; ++i)
#pragma unroll
      for (int j = 0; j < 4; ++j) acc[i][j] = zero;
  }

  union FB { uint4 u[2]; v16bf v; };

  auto compute = [&](int buf) {
    const unsigned short* aBase = &sA[buf][(wm * 64 + lr) * LDK + lkb];
    FB fb[4];
#pragma unroll
    for (int ni = 0; ni < 4; ++ni) {
      const unsigned short* p = &sB[buf][(wn * 64 + ni * 16 + lr) * LDK + lkb];
      fb[ni].u[0] = *(const uint4*)p;          // K = lkb .. lkb+7
      fb[ni].u[1] = *(const uint4*)(p + 16);   // K = lkb+16 .. lkb+23
    }
    // Software-pipelined A fragments: issue fa[mi+1] loads before WMMAs on fa[mi]
    FB fa[2];
    fa[0].u[0] = *(const uint4*)(aBase);
    fa[0].u[1] = *(const uint4*)(aBase + 16);
#pragma unroll
    for (int mi = 0; mi < 4; ++mi) {
      if (mi + 1 < 4) {
        const unsigned short* p = aBase + (mi + 1) * 16 * LDK;
        fa[(mi + 1) & 1].u[0] = *(const uint4*)p;
        fa[(mi + 1) & 1].u[1] = *(const uint4*)(p + 16);
      }
#pragma unroll
      for (int ni = 0; ni < 4; ++ni)
        acc[mi][ni] = __builtin_amdgcn_wmma_f32_16x16x32_bf16(
            false, fa[mi & 1].v, false, fb[ni].v, (short)0, acc[mi][ni], false, false);
    }
  };

#if USE_TDM
  // ---- TDM pipeline: wave 0 DMAs next tiles while all waves compute ----
  if (wid == 0) {
    tdm_load_tile(lds_off(&sA[0][0]), xb + (size_t)m0 * IN_F, BM);
    tdm_load_tile(lds_off(&sB[0][0]), wbase + (size_t)n0 * IN_F, BN);
    __builtin_amdgcn_s_wait_tensorcnt(0);
  }
  __syncthreads();

#pragma unroll 1   // keep live ranges within one iteration: <256 VGPRs, no msb/mov churn
  for (int kt = 0; kt < NK; ++kt) {
    const int cur = kt & 1;
    if (wid == 0 && kt + 1 < NK) {
      const int kb = (kt + 1) * BK;
      tdm_load_tile(lds_off(&sA[cur ^ 1][0]), xb + (size_t)m0 * IN_F + kb, BM);
      tdm_load_tile(lds_off(&sB[cur ^ 1][0]), wbase + (size_t)n0 * IN_F + kb, BN);
    }
    compute(cur);                                        // overlaps with TDM
    if (wid == 0 && kt + 1 < NK) __builtin_amdgcn_s_wait_tensorcnt(0);
    __syncthreads();
  }
#else
  // ---- Fallback: registers -> LDS (bf16, no conversion; low pressure) ----
  uint4 rA[2], rB[4];
  auto loadG = [&](int kt) {
    const int kb = kt * BK;
#pragma unroll
    for (int i = 0; i < 2; ++i) {
      int c = i * 256 + tid;                 // 512 chunks: 128 rows x 4 chunks
      int row = c >> 2, col = (c & 3) * 8;
      rA[i] = *(const uint4*)(xb + (size_t)(m0 + row) * IN_F + kb + col);
    }
#pragma unroll
    for (int i = 0; i < 4; ++i) {
      int c = i * 256 + tid;                 // 1024 chunks: 256 rows x 4 chunks
      int row = c >> 2, col = (c & 3) * 8;
      rB[i] = *(const uint4*)(wbase + (size_t)(n0 + row) * IN_F + kb + col);
    }
  };
  auto storeL = [&](int buf) {
#pragma unroll
    for (int i = 0; i < 2; ++i) {
      int c = i * 256 + tid;
      int row = c >> 2, col = (c & 3) * 8;
      *(uint4*)&sA[buf][row * LDK + col] = rA[i];
    }
#pragma unroll
    for (int i = 0; i < 4; ++i) {
      int c = i * 256 + tid;
      int row = c >> 2, col = (c & 3) * 8;
      *(uint4*)&sB[buf][row * LDK + col] = rB[i];
    }
  };

  loadG(0); storeL(0); loadG(1);
  __syncthreads();
#pragma unroll 1
  for (int kt = 0; kt < NK; ++kt) {
    const int cur = kt & 1;
    if (kt + 1 < NK) storeL(cur ^ 1);
    if (kt + 2 < NK) loadG(kt + 2);
    compute(cur);
    __syncthreads();
  }
#endif

  // Epilogue: lane L, vgpr g holds (M = g + 8*(L/16), N = L%16)
#pragma unroll
  for (int mi = 0; mi < 4; ++mi) {
    int row0 = m0 + wm * 64 + mi * 16 + (lane >> 4) * 8;
#pragma unroll
    for (int ni = 0; ni < 4; ++ni) {
      int n = n0 + wn * 64 + ni * 16 + lr;
      float bv = bias[n];
#pragma unroll
      for (int g = 0; g < 8; ++g)
        out[(size_t)(row0 + g) * OUT_F + n] = acc[mi][ni][g] + bv;
    }
  }
}

extern "C" void kernel_launch(void* const* d_in, const int* in_sizes, int n_in,
                              void* d_out, int out_size, void* d_ws, size_t ws_size,
                              hipStream_t stream) {
  const float* x    = (const float*)d_in[0];
  const float* W    = (const float*)d_in[1];
  const float* bias = (const float*)d_in[2];
  const float* A    = (const float*)d_in[3];
  const float* Bw   = (const float*)d_in[4];
  const int*   sid  = (const int*)d_in[5];
  float* out = (float*)d_out;

  unsigned short* weff = (unsigned short*)d_ws;                    // 16.8 MB
  unsigned short* xb   = weff + (size_t)NADAPT * OUT_F * IN_F;     // 67.1 MB

  // 1) Fold LoRA into bf16 effective weights.
  prep_weff<<<(NADAPT * OUT_F * (IN_F / 8)) / 256, 256, 0, stream>>>(W, A, Bw, weff);

  // 2) Convert activations to bf16 once (TDM-ready layout).
  cvt_x<<<(MTOT * IN_F / 8) / 256, 256, 0, stream>>>(x, xb);

  // 3) Fused GEMM: 32768x1024x1024 bf16 WMMA, TDM-staged tiles, f32 accumulate.
  dim3 grid(MTOT / BM, OUT_F / BN);   // 256 x 4
  lora_gemm<<<grid, 256, 0, stream>>>(xb, weff, bias, sid, out);
}
